// Model_8933531975744
// MI455X (gfx1250) — matrix operand order repaired and checked by reading, not compiled, not run on hardware
//
#include <hip/hip_runtime.h>
#include <hip/hip_bf16.h>
#include <math.h>

// ---------------------------------------------------------------------------
// Top-k sparse attention, MI455X (gfx1250, wave32, WMMA).
// Q (B,L,H,E)=(4,2048,8,64) f32, K (B,S,H,E), V (B,S,H,D), out (B,L,H,D) f32
// scores = QK^T per head; keep entries >= 614th-largest per row; softmax(scale*s); A@V
//
// Kernel 1 (convert_kv): one-shot f32->bf16 convert + swizzle of K and V into
//   d_ws, laid out in WMMA B-fragment order (amortized over 128 L-tiles/head).
// Kernel 2 (topk_attn): one workgroup (8 wave32s) = one 16-row L-tile per head.
// ---------------------------------------------------------------------------

#define BATCH 4
#define HEADS 8
#define LDIM  2048
#define SDIM  2048
#define EDIM  64
#define DDIM  64
#define KSEL  614          // max(1, int(2048*0.3))
#define SCALE 0.125f       // 1/sqrt(64)

#define WTS   1028         // padded bf16-pair row stride (dwords): conflict-free b128 reads

// workspace: Kb = bf16 K, [b][h][s][e/2] dwords ; Vsw = bf16 V pair-swizzled,
// [b][h][pblk][d][8] dwords with pblk = s/16, entry j = pack(V[s0+2j,d], V[s0+2j+1,d])
#define KB_DWORDS (BATCH * HEADS * SDIM * (EDIM / 2))   // 2M dwords = 8 MB
#define VS_DWORDS (BATCH * HEADS * (SDIM / 16) * DDIM * 8)

typedef __attribute__((ext_vector_type(16))) __bf16 v16bf;
typedef __attribute__((ext_vector_type(2)))  __bf16 v2bf;
typedef __attribute__((ext_vector_type(8)))  float  v8f;

union FragBF {
    v16bf v;
    unsigned u[8];
    uint4    q[2];
};

// f32 pair -> packed bf16 dword. Native cvt if present, else round-half-up +
// single v_perm_b32 byte-pack (3 VALU total).
static __device__ __forceinline__ unsigned pk_bf16(float a, float b) {
#if __has_builtin(__builtin_amdgcn_cvt_pk_bf16_f32)
    v2bf p = __builtin_amdgcn_cvt_pk_bf16_f32(a, b);
    return __builtin_bit_cast(unsigned, p);
#else
    unsigned ua = __float_as_uint(a) + 0x8000u;
    unsigned ub = __float_as_uint(b) + 0x8000u;
    return __builtin_amdgcn_perm(ub, ua, 0x07060302u);  // {ub[31:16], ua[31:16]}
#endif
}

// ------------------------- Kernel 1: convert + swizzle ----------------------
__global__ void __launch_bounds__(256)
convert_kv(const float* __restrict__ K, const float* __restrict__ V,
           unsigned* __restrict__ Kb, unsigned* __restrict__ Vsw) {
    const int i = blockIdx.x * 256 + threadIdx.x;
    if (i < KB_DWORDS) {
        // i = ((b*H+h)*2048 + s)*32 + e/2
        const int ep = i & 31;
        const int s  = (i >> 5) & 2047;
        const int hb = i >> 16;                     // b*HEADS + h
        const int h  = hb & 7, bb = hb >> 3;
        const float* kp = K + (((size_t)bb * SDIM + s) * HEADS + h) * EDIM + ep * 2;
        Kb[i] = pk_bf16(kp[0], kp[1]);
    } else {
        const int j  = i - KB_DWORDS;
        // j = (((b*H+h)*128 + pblk)*64 + d)*8 + jj
        const int jj   = j & 7;
        const int d    = (j >> 3) & 63;
        const int pblk = (j >> 9) & 127;
        const int hb   = j >> 16;
        const int h = hb & 7, bb = hb >> 3;
        const int s = (pblk * 8 + jj) * 2;
        const float* vp = V + (((size_t)bb * SDIM + s) * HEADS + h) * DDIM + d;
        Vsw[j] = pk_bf16(vp[0], vp[HEADS * DDIM]);
    }
}

// LDS layout (dynamic):
//   float  sc [16][2048]          131072 B  fp32 scores for the 16-row L-tile
//   u32    wt [16][WTS]            65792 B  bf16x2 softmax weights (padded rows)
//   float  acc[16][64]              4096 B  fp32 partial-output accumulator
//   float  thr[16], mx[16], rsum[16]
#define SC_BYTES   (16 * SDIM * 4)
#define WT_BYTES   (16 * WTS * 4)
#define ACC_BYTES  (16 * DDIM * 4)
#define SMEM_BYTES (SC_BYTES + WT_BYTES + ACC_BYTES + 3 * 16 * 4)

// ------------------------- Kernel 2: attention ------------------------------
__global__ void
__launch_bounds__(256, 1)
topk_attn_kernel(const float* __restrict__ Q,
                 const unsigned* __restrict__ Kb,
                 const unsigned* __restrict__ Vsw,
                 float* __restrict__ Out) {
    extern __shared__ char smem[];
    float*    sc   = (float*)smem;                                  // [16][2048]
    unsigned* wt   = (unsigned*)(smem + SC_BYTES);                  // [16][WTS]
    float*    acc  = (float*)(smem + SC_BYTES + WT_BYTES);          // [16][64]
    float*    thr  = (float*)(smem + SC_BYTES + WT_BYTES + ACC_BYTES);
    float*    mx   = thr + 16;
    float*    rsum = mx + 16;

    const int tile = blockIdx.x;
    const int l0   = (tile % (LDIM / 16)) * 16;
    const int h    = (tile / (LDIM / 16)) % HEADS;
    const int b    =  tile / ((LDIM / 16) * HEADS);

    const int lane = threadIdx.x & 31;
    const int wv   = threadIdx.x >> 5;      // 8 waves
    const int half = lane >> 4;             // 0: lanes 0-15, 1: lanes 16-31
    const int l16  = lane & 15;

    // ---------------- Phase 1: scores = Q K^T  (bf16 WMMA, fp32 acc) --------
    // A fragment (16x32, row l16 per lane): half0 K=e0+0..7 / e0+16..23,
    // half1 K=e0+8..15 / e0+24..31 (two bf16 per VGPR).
    FragBF aq[2];
    {
        const float* qrow = Q + (((size_t)b * LDIM + l0 + l16) * HEADS + h) * EDIM;
        #pragma unroll
        for (int ks = 0; ks < 2; ++ks) {
            const int e0 = ks * 32 + half * 8;
            float4 x0 = *(const float4*)(qrow + e0);
            float4 x1 = *(const float4*)(qrow + e0 + 4);
            float4 y0 = *(const float4*)(qrow + e0 + 16);
            float4 y1 = *(const float4*)(qrow + e0 + 20);
            aq[ks].u[0] = pk_bf16(x0.x, x0.y); aq[ks].u[1] = pk_bf16(x0.z, x0.w);
            aq[ks].u[2] = pk_bf16(x1.x, x1.y); aq[ks].u[3] = pk_bf16(x1.z, x1.w);
            aq[ks].u[4] = pk_bf16(y0.x, y0.y); aq[ks].u[5] = pk_bf16(y0.z, y0.w);
            aq[ks].u[6] = pk_bf16(y1.x, y1.y); aq[ks].u[7] = pk_bf16(y1.z, y1.w);
        }
    }

    // pre-swizzled bf16 K slice for this head (L2-resident, 256 KB)
    const unsigned* kbase = Kb + ((size_t)(b * HEADS + h) * SDIM) * (EDIM / 2);

    // each wave computes 16 s-tiles (256 columns of S)
    for (int t = wv * 16; t < wv * 16 + 16; ++t) {
        const int s0 = t * 16;
        // B fragment: lane column N=s0+l16; two 4-dword runs per ks (table order)
        const unsigned* kp = kbase + (size_t)(s0 + l16) * (EDIM / 2) + half * 4;
        __builtin_prefetch(kp + 16 * (EDIM / 2), 0, 0);   // global_prefetch_b8

        v8f c = {};
        #pragma unroll
        for (int ks = 0; ks < 2; ++ks) {
            FragBF bk;
            bk.q[0] = *(const uint4*)(kp + ks * 16);      // k = ks*32+half*8 + 0..7
            bk.q[1] = *(const uint4*)(kp + ks * 16 + 8);  // k = ks*32+16+half*8 + 0..7
            c = __builtin_amdgcn_wmma_f32_16x16x32_bf16(
                    false, aq[ks].v, false, bk.v, (short)0, c, false, false);
        }
        // C/D layout: VGPR j -> row (j + half*8), col l16
        #pragma unroll
        for (int j = 0; j < 8; ++j)
            sc[(j + half * 8) * SDIM + s0 + l16] = c[j];
    }
    __syncthreads();

    // ---------------- Phase 2: exact kth-largest per row (2 rows per wave) --
    for (int rr = 0; rr < 2; ++rr) {
        const int r = wv * 2 + rr;
        const float* rowp = sc + (size_t)r * SDIM;

        float m = -INFINITY;
        unsigned keys[64];                         // register-resident keys
        #pragma unroll
        for (int i = 0; i < 64; ++i) {
            float f = rowp[lane + 32 * i];
            m = fmaxf(m, f);
            unsigned u = __float_as_uint(f);       // order-preserving map
            keys[i] = (u >> 31) ? ~u : (u ^ 0x80000000u);
        }
        #pragma unroll
        for (int off = 16; off > 0; off >>= 1)
            m = fmaxf(m, __shfl_xor(m, off, 32));

        // max T such that count(key >= T) >= KSEL  ==  KSEL-th largest key
        unsigned T = 0;
        for (int bit = 31; bit >= 0; --bit) {
            const unsigned cand = T | (1u << bit);
            int cnt = 0;
            #pragma unroll
            for (int i = 0; i < 64; ++i) cnt += (keys[i] >= cand) ? 1 : 0;
            #pragma unroll
            for (int off = 16; off > 0; off >>= 1)
                cnt += __shfl_xor(cnt, off, 32);
            if (cnt >= KSEL) T = cand;
        }
        if (lane == 0) {
            unsigned ub = (T & 0x80000000u) ? (T ^ 0x80000000u) : ~T;
            thr[r] = __uint_as_float(ub);
            mx[r]  = m;
        }
    }
    __syncthreads();

    // ---------------- Phase 2.5: bf16 weights + row sums --------------------
    for (int rr = 0; rr < 2; ++rr) {
        const int r = wv * 2 + rr;
        const float* rowp = sc + (size_t)r * SDIM;
        const float th = thr[r], m = mx[r];
        float lsum = 0.f;
        for (int s2 = lane; s2 < SDIM / 2; s2 += 32) {
            float f0 = rowp[2 * s2], f1 = rowp[2 * s2 + 1];
            float w0 = (f0 >= th) ? __expf((f0 - m) * SCALE) : 0.f;
            float w1 = (f1 >= th) ? __expf((f1 - m) * SCALE) : 0.f;
            lsum += w0 + w1;
            wt[r * WTS + s2] = pk_bf16(w0, w1);
        }
        #pragma unroll
        for (int off = 16; off > 0; off >>= 1)
            lsum += __shfl_xor(lsum, off, 32);
        if (lane == 0) rsum[r] = lsum;
    }
    // zero the output accumulator (distinct LDS region)
    for (int i = threadIdx.x; i < 16 * DDIM; i += 256) acc[i] = 0.f;
    __syncthreads();

    // ---------------- Phase 3: out = (W @ V) / rowsum  ----------------------
    // wave pair (w, w+4) shares D-tile (w&3); each covers half of S.
    const int d0   = (wv & 3) * 16;
    const int sbeg = (wv >> 2) * (SDIM / 2);
    const unsigned* vbase = Vsw + (size_t)(b * HEADS + h) * (SDIM / 16) * DDIM * 8;

    v8f c = {};
    for (int s0 = sbeg; s0 < sbeg + SDIM / 2; s0 += 32) {
        // A fragment: row l16, k pairs from wt (LDS), contiguous k order (16*half + 0..15)
        const unsigned* wrow = wt + l16 * WTS;
        const int base = s0 / 2 + half * 8;
        FragBF aw;
        aw.q[0] = *(const uint4*)(wrow + base);       // k = s0 + half*16 .. +7
        aw.q[1] = *(const uint4*)(wrow + base + 4);   // k = s0+half*16+8 .. +7

        // B fragment: lane column d=d0+l16, k = s0+half*16+0..15; pre-swizzled
        const unsigned* vp = vbase + ((size_t)(s0 / 16 + half) * DDIM + d0 + l16) * 8;
        FragBF bv;
        bv.q[0] = *(const uint4*)(vp);
        bv.q[1] = *(const uint4*)(vp + 4);

        c = __builtin_amdgcn_wmma_f32_16x16x32_bf16(
                false, aw.v, false, bv.v, (short)0, c, false, false);
    }
    // combine the two S-halves with LDS float atomics (ds_add_f32)
    #pragma unroll
    for (int j = 0; j < 8; ++j) {
        const int row = j + half * 8;
        atomicAdd(&acc[row * DDIM + d0 + l16], c[j]);
    }
    __syncthreads();

    // coalesced normalize + store
    for (int i = threadIdx.x; i < 16 * DDIM; i += 256) {
        const int row = i >> 6;
        const int d   = i & 63;
        Out[(((size_t)b * LDIM + l0 + row) * HEADS + h) * DDIM + d] = acc[i] / rsum[row];
    }
}

extern "C" void kernel_launch(void* const* d_in, const int* in_sizes, int n_in,
                              void* d_out, int out_size, void* d_ws, size_t ws_size,
                              hipStream_t stream) {
    const float* Q = (const float*)d_in[0];
    const float* K = (const float*)d_in[1];
    const float* V = (const float*)d_in[2];
    float* Out = (float*)d_out;

    (void)in_sizes; (void)n_in; (void)out_size; (void)ws_size;

    unsigned* Kb  = (unsigned*)d_ws;                  // 8 MB
    unsigned* Vsw = (unsigned*)d_ws + KB_DWORDS;      // 8 MB

    hipFuncSetAttribute((const void*)topk_attn_kernel,
                        hipFuncAttributeMaxDynamicSharedMemorySize, SMEM_BYTES);

    const int cvt_threads = KB_DWORDS + VS_DWORDS;    // 4M dwords
    convert_kv<<<cvt_threads / 256, 256, 0, stream>>>(K, V, Kb, Vsw);

    const int grid = BATCH * HEADS * (LDIM / 16);     // 4096 workgroups
    topk_attn_kernel<<<grid, 256, SMEM_BYTES, stream>>>(Q, Kb, Vsw, Out);
}
